// RVSAEncoder_52682068853185
// MI455X (gfx1250) — matrix operand order repaired and checked by reading, not compiled, not run on hardware
//
#include <hip/hip_runtime.h>
#include <math.h>

typedef __attribute__((ext_vector_type(16))) __bf16 bf16x16;
typedef __attribute__((ext_vector_type(8)))  __bf16 bf16x8;
typedef __attribute__((ext_vector_type(8)))  float  v8f;

static constexpr float kBnScale = 0.9999950000374997f; // 1/sqrt(1+1e-5)

// A fragment (16x32 bf16, MxK): lane holds row m=l16; elems 0..7 = K[k0..k0+7],
// elems 8..15 = K[k0+16..k0+23], where k0 = kc*32 + half*8  (ISA 7.12.2 layout)
__device__ __forceinline__ bf16x16 load_a_frag(const __bf16* row, int k0) {
  bf16x8 lo = *(const bf16x8*)(row + k0);
  bf16x8 hi = *(const bf16x8*)(row + k0 + 16);
  return __builtin_shufflevector(lo, hi, 0,1,2,3,4,5,6,7,8,9,10,11,12,13,14,15);
}
// B fragment (32x16 bf16, KxN): lane holds col n=l16; elems 0..7 = K[k0..k0+7],
// elems 8..15 = K[k0+16..k0+23], where k0 = kc*32 + half*8
__device__ __forceinline__ bf16x16 load_b_frag(const __bf16* row, int k0) {
  bf16x8 lo = *(const bf16x8*)(row + k0);
  bf16x8 hi = *(const bf16x8*)(row + k0 + 16);
  return __builtin_shufflevector(lo, hi, 0,1,2,3,4,5,6,7,8,9,10,11,12,13,14,15);
}
__device__ __forceinline__ v8f wmma_bf16(bf16x16 a, bf16x16 b, v8f c) {
  return __builtin_amdgcn_wmma_f32_16x16x32_bf16(false, a, false, b, (short)0, c,
                                                 false, false);
}
__device__ __forceinline__ float xor16_max(float x) {
  float o = __int_as_float(__builtin_amdgcn_ds_swizzle(__float_as_int(x), 0x401f)); // SWAPX16
  return fmaxf(x, o);
}

// ---------------------------------------------------------------------------
// Kernel 1: per src point, write rec_l[n] = { A_src_l[64], cur_l[64] } (fp32).
// cur_l via WMMA GEMM (128 rows/block), A_src_l via VALU (3-wide K).
// ---------------------------------------------------------------------------
__global__ __launch_bounds__(256) void k_precompute(
    const float* __restrict__ src_feats, const float* __restrict__ src_xyz,
    const float* __restrict__ fc0_w, const float* __restrict__ fc0_b,
    const float* __restrict__ p0_w,  const float* __restrict__ p0_g,
    const float* __restrict__ fc1_w, const float* __restrict__ fc1_b,
    const float* __restrict__ p1_w,  const float* __restrict__ p1_g,
    float* __restrict__ rec0, float* __restrict__ rec1)
{
  __shared__ __attribute__((aligned(16))) __bf16 a_lds[128][72];
  __shared__ __attribute__((aligned(16))) __bf16 w_lds[2][64][72];

  const int tid = threadIdx.x;
  const int lane = tid & 31, wave = tid >> 5;
  const int half = lane >> 4, l16 = lane & 15;
  const int pbase = blockIdx.x * 128;

  for (int i = tid; i < 128 * 64; i += 256)
    a_lds[i >> 6][i & 63] = (__bf16)src_feats[(size_t)(pbase + (i >> 6)) * 64 + (i & 63)];
  for (int i = tid; i < 64 * 64; i += 256) {
    w_lds[0][i >> 6][i & 63] = (__bf16)fc0_w[i];
    w_lds[1][i >> 6][i & 63] = (__bf16)fc1_w[i];
  }

  { // A_src: thread handles one (point, layer): 64 outputs, 3 FMA each
    int p = tid & 127, layer = tid >> 7;
    const float* pw = layer ? p1_w : p0_w;
    const float* pg = layer ? p1_g : p0_g;
    float* rec = layer ? rec1 : rec0;
    int gp = pbase + p;
    float x = src_xyz[gp * 3 + 0], y = src_xyz[gp * 3 + 1], z = src_xyz[gp * 3 + 2];
    float* out = rec + (size_t)gp * 128;
    for (int d = 0; d < 64; ++d) {
      float gs = pg[d] * kBnScale;
      out[d] = (x * pw[d * 6 + 0] + y * pw[d * 6 + 1] + z * pw[d * 6 + 2]) * gs;
    }
  }
  __syncthreads();

  // cur GEMM: 2 layers x 8 row-tiles x 4 n-tiles = 64 tiles over 8 waves
  for (int q = 0; q < 8; ++q) {
    int T = q * 8 + wave;
    int layer = T >> 5, rt = (T >> 2) & 7, nt = T & 3;
    int m = rt * 16 + l16, n = nt * 16 + l16;
    v8f acc = {};
#pragma unroll
    for (int kc = 0; kc < 2; ++kc) {
      bf16x16 a = load_a_frag(&a_lds[m][0], kc * 32 + half * 8);
      bf16x16 b = load_b_frag(&w_lds[layer][n][0], kc * 32 + half * 8);
      acc = wmma_bf16(a, b, acc);
    }
    const float* fb = layer ? fc1_b : fc0_b;
    float* rec = layer ? rec1 : rec0;
    float bias = fb[n];
#pragma unroll
    for (int v = 0; v < 8; ++v) {
      int row = rt * 16 + v + 8 * half;
      rec[(size_t)(pbase + row) * 128 + 64 + n] = acc[v] + bias;
    }
  }
}

// ---------------------------------------------------------------------------
// Kernel 2: SA layer. 256 gathered rows per block (256/S dst points).
// x = silu(A_src[idx] + A_dst[m]) + cur[idx] (zeroed when empty) -> bf16 LDS,
// WMMA GEMM vs m_w (H=128), bn+relu, max over S, write feats[:, layerOff..].
// ---------------------------------------------------------------------------
template <int S>
__global__ __launch_bounds__(256) void k_sa(
    const float* __restrict__ rec, const float* __restrict__ dst_xyz,
    const int* __restrict__ idx, const unsigned char* __restrict__ empty,
    const float* __restrict__ p_w, const float* __restrict__ p_g,
    const float* __restrict__ p_b, const float* __restrict__ m_w,
    const float* __restrict__ m_g, const float* __restrict__ m_b,
    float* __restrict__ feats, int layerOff)
{
  __shared__ __attribute__((aligned(16))) __bf16 x_lds[256][72];
  __shared__ __attribute__((aligned(16))) __bf16 w_lds[128][72];
  __shared__ float pwe[64][4];
  __shared__ float mgb[128][2];

  const int tid = threadIdx.x;
  const int lane = tid & 31, wave = tid >> 5;
  const int half = lane >> 4, l16 = lane & 15;
  const int dstBase = blockIdx.x * (256 / S);

  for (int i = tid; i < 128 * 64; i += 256)
    w_lds[i >> 6][i & 63] = (__bf16)m_w[i];
  if (tid < 128) { mgb[tid][0] = m_g[tid] * kBnScale; mgb[tid][1] = m_b[tid]; }
  if (tid < 64) {
    float gs = p_g[tid] * kBnScale;
    pwe[tid][0] = (p_w[tid * 6 + 3] - p_w[tid * 6 + 0]) * gs;
    pwe[tid][1] = (p_w[tid * 6 + 4] - p_w[tid * 6 + 1]) * gs;
    pwe[tid][2] = (p_w[tid * 6 + 5] - p_w[tid * 6 + 2]) * gs;
    pwe[tid][3] = p_b[tid];
  }
  __syncthreads();

  { // build one gathered row per thread
    int r = tid;
    int gr = blockIdx.x * 256 + r;
    int gm = gr / S, s = gr % S;
    int srcI = idx[(size_t)gm * S + s];
    bool em = empty[gm] != 0;
    float dx = dst_xyz[gm * 3 + 0], dy = dst_xyz[gm * 3 + 1], dz = dst_xyz[gm * 3 + 2];
    const float* rp = rec + (size_t)srcI * 128;
#pragma unroll
    for (int d = 0; d < 64; d += 4) {
      float4 a4 = *(const float4*)(rp + d);
      float4 c4 = *(const float4*)(rp + 64 + d);
      float az[4] = {a4.x, a4.y, a4.z, a4.w};
      float cz[4] = {c4.x, c4.y, c4.z, c4.w};
#pragma unroll
      for (int j = 0; j < 4; ++j) {
        int dd = d + j;
        float z = az[j] + dx * pwe[dd][0] + dy * pwe[dd][1] + dz * pwe[dd][2] + pwe[dd][3];
        float sil = z / (1.0f + __expf(-z));
        float xv = sil + cz[j];
        x_lds[r][dd] = (__bf16)(em ? 0.0f : xv);
      }
    }
  }
  __syncthreads();

  // wave w owns row-tiles {2w, 2w+1}; S==32 -> same dst (combine max), S==16 -> 2 dsts
  for (int nt = 0; nt < 8; ++nt) {
    float red[2];
#pragma unroll
    for (int rtl = 0; rtl < 2; ++rtl) {
      int rt = wave * 2 + rtl;
      v8f acc = {};
#pragma unroll
      for (int kc = 0; kc < 2; ++kc) {
        bf16x16 a = load_a_frag(&x_lds[rt * 16 + l16][0], kc * 32 + half * 8);
        bf16x16 b = load_b_frag(&w_lds[nt * 16 + l16][0], kc * 32 + half * 8);
        acc = wmma_bf16(a, b, acc);
      }
      int col = nt * 16 + l16;
      float gs = mgb[col][0], bb = mgb[col][1];
      float mx = 0.0f; // relu output >= 0, so 0 is a safe identity
#pragma unroll
      for (int v = 0; v < 8; ++v)
        mx = fmaxf(mx, fmaxf(acc[v] * gs + bb, 0.0f));
      red[rtl] = xor16_max(mx); // fold the two 8-row halves
    }
    int col = nt * 16 + l16;
    if (lane < 16) {
      if (S == 32) {
        feats[(size_t)(dstBase + wave) * 256 + layerOff + col] = fmaxf(red[0], red[1]);
      } else {
        feats[(size_t)(dstBase + wave * 2 + 0) * 256 + layerOff + col] = red[0];
        feats[(size_t)(dstBase + wave * 2 + 1) * 256 + layerOff + col] = red[1];
      }
    }
  }
}

// ---------------------------------------------------------------------------
// Kernel 3: out = relu(bn(feats @ out_w^T)).  64 rows x 256 cols per block,
// K=256 in 4 LDS chunks of 64.
// ---------------------------------------------------------------------------
__global__ __launch_bounds__(256) void k_out(
    const float* __restrict__ feats, const float* __restrict__ out_w,
    const float* __restrict__ out_g, const float* __restrict__ out_b,
    float* __restrict__ out)
{
  __shared__ __attribute__((aligned(16))) __bf16 a_lds[64][72];
  __shared__ __attribute__((aligned(16))) __bf16 b_lds[256][72];
  __shared__ float ogb[256][2];

  const int tid = threadIdx.x;
  const int lane = tid & 31, wave = tid >> 5;
  const int half = lane >> 4, l16 = lane & 15;
  const int m0 = blockIdx.x * 64;
  const int rt = wave >> 1;          // 4 row tiles, 2 waves each
  const int nb = (wave & 1) * 8;     // 8 n-tiles per wave

  ogb[tid][0] = out_g[tid] * kBnScale;
  ogb[tid][1] = out_b[tid];

  v8f acc[8] = {};
  for (int kc = 0; kc < 4; ++kc) {
    __syncthreads();
    for (int i = tid; i < 64 * 64; i += 256)
      a_lds[i >> 6][i & 63] =
          (__bf16)feats[(size_t)(m0 + (i >> 6)) * 256 + kc * 64 + (i & 63)];
    for (int i = tid; i < 256 * 64; i += 256)
      b_lds[i >> 6][i & 63] = (__bf16)out_w[(size_t)(i >> 6) * 256 + kc * 64 + (i & 63)];
    __syncthreads();
#pragma unroll
    for (int j = 0; j < 8; ++j) {
      int nt = nb + j;
#pragma unroll
      for (int c = 0; c < 2; ++c) {
        bf16x16 a = load_a_frag(&a_lds[rt * 16 + l16][0], c * 32 + half * 8);
        bf16x16 b = load_b_frag(&b_lds[nt * 16 + l16][0], c * 32 + half * 8);
        acc[j] = wmma_bf16(a, b, acc[j]);
      }
    }
  }
#pragma unroll
  for (int j = 0; j < 8; ++j) {
    int col = (nb + j) * 16 + l16;
    float gs = ogb[col][0], bb = ogb[col][1];
#pragma unroll
    for (int v = 0; v < 8; ++v) {
      int row = m0 + rt * 16 + v + 8 * half;
      out[(size_t)row * 256 + col] = fmaxf(acc[j][v] * gs + bb, 0.0f);
    }
  }
}

extern "C" void kernel_launch(void* const* d_in, const int* in_sizes, int n_in,
                              void* d_out, int out_size, void* d_ws, size_t ws_size,
                              hipStream_t stream) {
  const float* src_xyz   = (const float*)d_in[0];
  const float* src_feats = (const float*)d_in[1];
  const float* dst_xyz   = (const float*)d_in[2];
  const int*   idx0      = (const int*)d_in[3];
  const int*   idx1      = (const int*)d_in[4];
  const unsigned char* empty0 = (const unsigned char*)d_in[5];
  const unsigned char* empty1 = (const unsigned char*)d_in[6];
  const float* fc0_w = (const float*)d_in[7];
  const float* fc0_b = (const float*)d_in[8];
  const float* p0_w  = (const float*)d_in[9];
  const float* p0_g  = (const float*)d_in[10];
  const float* p0_b  = (const float*)d_in[11];
  const float* m0_w  = (const float*)d_in[12];
  const float* m0_g  = (const float*)d_in[13];
  const float* m0_b  = (const float*)d_in[14];
  const float* fc1_w = (const float*)d_in[15];
  const float* fc1_b = (const float*)d_in[16];
  const float* p1_w  = (const float*)d_in[17];
  const float* p1_g  = (const float*)d_in[18];
  const float* p1_b  = (const float*)d_in[19];
  const float* m1_w  = (const float*)d_in[20];
  const float* m1_g  = (const float*)d_in[21];
  const float* m1_b  = (const float*)d_in[22];
  const float* out_w = (const float*)d_in[23];
  const float* out_g = (const float*)d_in[24];
  const float* out_b = (const float*)d_in[25];

  const int N = 65536, M = 16384;
  float* rec0  = (float*)d_ws;                 // N x 128: [A_src0 | cur0]
  float* rec1  = rec0 + (size_t)N * 128;       // N x 128: [A_src1 | cur1]
  float* feats = rec1 + (size_t)N * 128;       // M x 256 concat features

  k_precompute<<<N / 128, 256, 0, stream>>>(src_feats, src_xyz, fc0_w, fc0_b, p0_w,
                                            p0_g, fc1_w, fc1_b, p1_w, p1_g, rec0, rec1);
  k_sa<16><<<(M * 16) / 256, 256, 0, stream>>>(rec0, dst_xyz, idx0, empty0, p0_w, p0_g,
                                               p0_b, m0_w, m0_g, m0_b, feats, 0);
  k_sa<32><<<(M * 32) / 256, 256, 0, stream>>>(rec1, dst_xyz, idx1, empty1, p1_w, p1_g,
                                               p1_b, m1_w, m1_g, m1_b, feats, 128);
  k_out<<<M / 64, 256, 0, stream>>>(feats, out_w, out_g, out_b, (float*)d_out);
}
